// LazyEqProp_46084999086194
// MI455X (gfx1250) — matrix operand order repaired and checked by reading, not compiled, not run on hardware
//
#include <hip/hip_runtime.h>
#include <hip/hip_bf16.h>
#include <math.h>

typedef __attribute__((ext_vector_type(16))) __bf16 v16bf;
typedef __attribute__((ext_vector_type(8)))  float  v8f;

// ---------------------------------------------------------------------------
// Utility kernels
// ---------------------------------------------------------------------------
__global__ void cvt_f32_bf16(const float* __restrict__ in, __bf16* __restrict__ out, size_t n) {
    size_t i = (size_t)blockIdx.x * blockDim.x + threadIdx.x;
    size_t stride = (size_t)gridDim.x * blockDim.x;
    for (; i < n; i += stride) out[i] = (__bf16)in[i];
}

__global__ void transpose_cvt_bf16(const float* __restrict__ in, __bf16* __restrict__ outT,
                                   int R, int C) { // in[R][C] -> outT[C][R]
    size_t n = (size_t)R * C;
    size_t i = (size_t)blockIdx.x * blockDim.x + threadIdx.x;
    size_t stride = (size_t)gridDim.x * blockDim.x;
    for (; i < n; i += stride) {
        int r = (int)(i / C), c = (int)(i % C);
        outT[(size_t)c * R + r] = (__bf16)in[i];
    }
}

__global__ void fill_zero_u16(unsigned short* __restrict__ p, size_t n) {
    size_t i = (size_t)blockIdx.x * blockDim.x + threadIdx.x;
    size_t stride = (size_t)gridDim.x * blockDim.x;
    for (; i < n; i += stride) p[i] = 0;
}

// ---------------------------------------------------------------------------
// Softmax over O, then subtract mean (mean of softmax row == 1/O exactly).
// One block per row.
// ---------------------------------------------------------------------------
__global__ void softmax_center(const float* __restrict__ logits, __bf16* __restrict__ P, int O) {
    const int row = blockIdx.x, t = threadIdx.x;
    const float* Lr = logits + (size_t)row * O;
    __shared__ float red[256];
    float mx = -INFINITY;
    for (int j = t; j < O; j += 256) mx = fmaxf(mx, Lr[j]);
    red[t] = mx; __syncthreads();
    for (int s = 128; s > 0; s >>= 1) { if (t < s) red[t] = fmaxf(red[t], red[t + s]); __syncthreads(); }
    mx = red[0]; __syncthreads();
    float sm = 0.0f;
    for (int j = t; j < O; j += 256) sm += __expf(Lr[j] - mx);
    red[t] = sm; __syncthreads();
    for (int s = 128; s > 0; s >>= 1) { if (t < s) red[t] += red[t + s]; __syncthreads(); }
    const float inv = 1.0f / red[0];
    const float c = 1.0f / (float)O;
    for (int j = t; j < O; j += 256)
        P[(size_t)row * O + j] = (__bf16)(__expf(Lr[j] - mx) * inv - c);
}

// ---------------------------------------------------------------------------
// Per-row lazy-update gate: inp = src - nscale*nudge ; delta = mean|inp - prev|
// mask[row] = delta > EPS ; prev[row] = inp.  One block per row.
// ---------------------------------------------------------------------------
__global__ void mask_prev(const __bf16* __restrict__ src, const __bf16* __restrict__ nudge,
                          float nscale, __bf16* __restrict__ prev, float* __restrict__ mask,
                          int Hd) {
    const int row = blockIdx.x, t = threadIdx.x;
    const size_t base = (size_t)row * Hd;
    __shared__ float red[256];
    float s = 0.0f;
    for (int j = t; j < Hd; j += 256) {
        float v = (float)src[base + j];
        if (nudge) v -= nscale * (float)nudge[base + j];
        s += fabsf(v - (float)prev[base + j]);
        prev[base + j] = (__bf16)v;
    }
    red[t] = s; __syncthreads();
    for (int st = 128; st > 0; st >>= 1) { if (t < st) red[t] += red[t + st]; __syncthreads(); }
    if (t == 0) mask[row] = (red[0] / (float)Hd > 0.01f) ? 1.0f : 0.0f;
}

// ---------------------------------------------------------------------------
// WMMA bf16 GEMM: C[M,N] = A[M,K] @ W[N,K]^T (+bias), f32 accumulate.
// Block 256 threads = 8 wave32, block tile 128x128, wave tile 32x64 (2x4 WMMAs).
// Global->LDS staging via GLOBAL_LOAD_ASYNC_TO_LDS_B128 (ASYNCcnt path).
// mode 0: write f32      mode 2: write bf16
// mode 1: h = mask ? 0.5*h_old + 0.5*tanh(acc+bias) : h_old   (in-place bf16)
// ---------------------------------------------------------------------------
union FragU { uint4 u[2]; v16bf v; };

__global__ __launch_bounds__(256) void wmma_gemm_bf16(
    const __bf16* __restrict__ A, const __bf16* __restrict__ W,
    const float* __restrict__ bias, void* __restrict__ Cout,
    const float* __restrict__ mask, int M, int N, int K, int mode) {

    __shared__ alignas(16) __bf16 lA[128][32];
    __shared__ alignas(16) __bf16 lW[128][32];

    const int t     = threadIdx.x;
    const int lane  = t & 31;
    const int wv    = t >> 5;            // 0..7
    const int wm0   = (wv >> 1) * 32;    // 0,32,64,96
    const int wn0   = (wv & 1) * 64;     // 0,64
    const int lrow  = lane & 15;
    const int lhalf = lane >> 4;         // K-group select per ISA VGPR layout
    const int m0 = blockIdx.y * 128, n0 = blockIdx.x * 128;
    const int r  = t >> 1, ko = (t & 1) * 16;   // LDS fill: 2 threads/row, 16 bf16 each

    const __bf16* gA = A + (size_t)(m0 + r) * K + ko;
    const __bf16* gW = W + (size_t)(n0 + r) * K + ko;

    // Raw LDS byte offsets (low 32 bits of generic shared pointer == LDS offset)
    const unsigned ldsA = (unsigned)(size_t)(&lA[r][ko]);
    const unsigned ldsW = (unsigned)(size_t)(&lW[r][ko]);

    v8f acc[2][4];
    #pragma unroll
    for (int i = 0; i < 2; ++i)
        #pragma unroll
        for (int j = 0; j < 4; ++j)
            #pragma unroll
            for (int e = 0; e < 8; ++e) acc[i][j][e] = 0.0f;

    for (int k0 = 0; k0 < K; k0 += 32) {
        // Async copy global -> LDS: 32B per thread per tile, INST_OFFSET applies
        // to both LDS and global addresses (ISA 10.x / 15.18).
        {
            const __bf16* pA = gA + k0;
            const __bf16* pW = gW + k0;
            asm volatile(
                "global_load_async_to_lds_b128 %0, %2, off\n\t"
                "global_load_async_to_lds_b128 %0, %2, off offset:16\n\t"
                "global_load_async_to_lds_b128 %1, %3, off\n\t"
                "global_load_async_to_lds_b128 %1, %3, off offset:16"
                :: "v"(ldsA), "v"(ldsW), "v"(pA), "v"(pW)
                : "memory");
        }
        if (k0 + 32 < K) {          // global_prefetch_b8 for next K slab
            __builtin_prefetch(gA + k0 + 32, 0, 1);
            __builtin_prefetch(gW + k0 + 32, 0, 1);
        }
        asm volatile("s_wait_asynccnt 0" ::: "memory");
        __syncthreads();

        // A frag (16x32 bf16): lane half 0 -> K 0-15, half 1 -> K 16-31
        FragU af[2];
        #pragma unroll
        for (int tm = 0; tm < 2; ++tm) {
            const __bf16* p = &lA[wm0 + tm * 16 + lrow][lhalf * 16];
            af[tm].u[0] = *(const uint4*)p;
            af[tm].u[1] = *(const uint4*)(p + 8);
        }
        // B frag (32x16 bf16): lane half 0 -> K 0-15, half 1 -> K 16-31 (contig row of W)
        FragU bf_[4];
        #pragma unroll
        for (int tn = 0; tn < 4; ++tn) {
            const __bf16* p = &lW[wn0 + tn * 16 + lrow][lhalf * 16];
            bf_[tn].u[0] = ((const uint4*)p)[0];
            bf_[tn].u[1] = ((const uint4*)p)[1];
        }
        #pragma unroll
        for (int tm = 0; tm < 2; ++tm)
            #pragma unroll
            for (int tn = 0; tn < 4; ++tn)
                acc[tm][tn] = __builtin_amdgcn_wmma_f32_16x16x32_bf16(
                    false, af[tm].v, false, bf_[tn].v, (short)0, acc[tm][tn], false, false);
        __syncthreads();
    }

    // Epilogue. C/D layout: VGPR v -> row v + 8*lhalf, col = lrow.
    #pragma unroll
    for (int tm = 0; tm < 2; ++tm)
        #pragma unroll
        for (int tn = 0; tn < 4; ++tn) {
            const int gn = n0 + wn0 + tn * 16 + lrow;
            const float bv = bias ? bias[gn] : 0.0f;
            #pragma unroll
            for (int v = 0; v < 8; ++v) {
                const int gm = m0 + wm0 + tm * 16 + lhalf * 8 + v;
                const size_t idx = (size_t)gm * N + gn;
                const float a = acc[tm][tn][v] + bv;
                if (mode == 0) {
                    ((float*)Cout)[idx] = a;
                } else if (mode == 2) {
                    ((__bf16*)Cout)[idx] = (__bf16)a;
                } else {
                    __bf16* Hp = (__bf16*)Cout;
                    const float hold = (float)Hp[idx];
                    const float hu = 0.5f * hold + 0.5f * tanhf(a);   // ALPHA = 0.5
                    const float mk = mask[gm];
                    Hp[idx] = (__bf16)(mk * hu + (1.0f - mk) * hold);
                }
            }
        }
}

// ---------------------------------------------------------------------------
// Host orchestration
// ---------------------------------------------------------------------------
extern "C" void kernel_launch(void* const* d_in, const int* in_sizes, int n_in,
                              void* d_out, int out_size, void* d_ws, size_t ws_size,
                              hipStream_t stream) {
    constexpr int Bn = 4096, In = 1024, Hn = 2048, On = 1024, Ln = 3, STEPS = 30;
    const float* x        = (const float*)d_in[0];
    const float* W_embed  = (const float*)d_in[1];
    const float* b_embed  = (const float*)d_in[2];
    const float* W_layers = (const float*)d_in[3];
    const float* b_layers = (const float*)d_in[4];
    const float* W_head   = (const float*)d_in[5];
    const float* b_head   = (const float*)d_in[6];
    float* outp = (float*)d_out;

    char* ws = (char*)d_ws;
    size_t off = 0;
    auto take = [&](size_t bytes) -> char* {
        char* p = ws + off;
        off += (bytes + 255) & ~(size_t)255;
        return p;
    };
    __bf16* xb   = (__bf16*)take((size_t)Bn * In * 2);
    __bf16* Web  = (__bf16*)take((size_t)Hn * In * 2);
    __bf16* Wlb  = (__bf16*)take((size_t)Ln * Hn * Hn * 2);
    __bf16* Whb  = (__bf16*)take((size_t)On * Hn * 2);
    __bf16* WhTb = (__bf16*)take((size_t)Hn * On * 2);
    __bf16* xemb = (__bf16*)take((size_t)Bn * Hn * 2);
    __bf16* h    = (__bf16*)take((size_t)Ln * Bn * Hn * 2);
    __bf16* prev = (__bf16*)take((size_t)Ln * Bn * Hn * 2);   // contiguous after h
    __bf16* nud  = (__bf16*)take((size_t)Bn * Hn * 2);
    __bf16* P    = (__bf16*)take((size_t)Bn * On * 2);
    float*  logi = (float*)take((size_t)Bn * On * 4);
    float*  mask = (float*)take((size_t)Ln * Bn * 4);

    const dim3 blk(256);

    // --- precision conversion (every call; no cached state allowed) ---
    cvt_f32_bf16<<<2048, blk, 0, stream>>>(x,        xb,  (size_t)Bn * In);
    cvt_f32_bf16<<<2048, blk, 0, stream>>>(W_embed,  Web, (size_t)Hn * In);
    cvt_f32_bf16<<<4096, blk, 0, stream>>>(W_layers, Wlb, (size_t)Ln * Hn * Hn);
    cvt_f32_bf16<<<2048, blk, 0, stream>>>(W_head,   Whb, (size_t)On * Hn);
    transpose_cvt_bf16<<<2048, blk, 0, stream>>>(W_head, WhTb, On, Hn);
    fill_zero_u16<<<4096, blk, 0, stream>>>((unsigned short*)h, (size_t)2 * Ln * Bn * Hn);

    // x_emb = x @ W_embed^T + b_embed  -> bf16
    wmma_gemm_bf16<<<dim3(Hn / 128, Bn / 128), blk, 0, stream>>>(
        xb, Web, b_embed, xemb, nullptr, Bn, Hn, In, 2);

    for (int step = 0; step < STEPS; ++step) {
        // logits = h[2] @ W_head^T + b_head (f32)
        wmma_gemm_bf16<<<dim3(On / 128, Bn / 128), blk, 0, stream>>>(
            h + (size_t)2 * Bn * Hn, Whb, b_head, logi, nullptr, Bn, On, Hn, 0);
        // P = softmax(logits) - 1/O
        softmax_center<<<Bn, blk, 0, stream>>>(logi, P, On);
        // nudge = P @ W_head  (== P @ WhT^T)
        wmma_gemm_bf16<<<dim3(Hn / 128, Bn / 128), blk, 0, stream>>>(
            P, WhTb, nullptr, nud, nullptr, Bn, Hn, On, 2);

        // gates + prev updates (all read OLD h / prev; must precede layer GEMMs)
        mask_prev<<<Bn, blk, 0, stream>>>(xemb, nullptr, 0.0f,
                                          prev,                      mask,          Hn);
        mask_prev<<<Bn, blk, 0, stream>>>(h, nullptr, 0.0f,
                                          prev + (size_t)Bn * Hn,    mask + Bn,     Hn);
        mask_prev<<<Bn, blk, 0, stream>>>(h + (size_t)Bn * Hn, nud,
                                          (step > 0) ? 0.1f : 0.0f,  // NUDGE gated by step>0
                                          prev + (size_t)2 * Bn * Hn, mask + 2 * Bn, Hn);

        // layer updates: A = prev[i] (== inp), fused tanh + alpha-blend + lazy mask
        for (int i = 0; i < Ln; ++i)
            wmma_gemm_bf16<<<dim3(Hn / 128, Bn / 128), blk, 0, stream>>>(
                prev + (size_t)i * Bn * Hn, Wlb + (size_t)i * Hn * Hn,
                b_layers + i * Hn, h + (size_t)i * Bn * Hn, mask + i * Bn,
                Bn, Hn, Hn, 1);
    }

    // out = h[2] @ W_head^T + b_head (f32)
    wmma_gemm_bf16<<<dim3(On / 128, Bn / 128), blk, 0, stream>>>(
        h + (size_t)2 * Bn * Hn, Whb, b_head, outp, nullptr, Bn, On, Hn, 0);
}
